// MultiHeadDiffAttention_1219770712320
// MI455X (gfx1250) — matrix operand order repaired and checked by reading, not compiled, not run on hardware
//
#include <hip/hip_runtime.h>
#include <hip/hip_bf16.h>

// ---------------------------------------------------------------------------
// Differential multi-head attention (Diff Transformer layer) for gfx1250.
// - all matmuls on v_wmma_f32_16x16x32_bf16 (bf16 operands, f32 accumulate)
// - operand tiles staged into LDS by the Tensor Data Mover
//   (tensor_load_to_lds, TENSORcnt), double-buffered to overlap DMA & WMMA
// - bf16 fragments read as 16B vectors (ds_load_b128 / global_load_b128)
// - softmax row reductions via DPP cross-lane moves (no LDS round-trips)
// ---------------------------------------------------------------------------

#define DIMM   1024
#define NHEAD  16
#define HD     64
#define SEQ    2048
#define BATCH  2
#define NROW   (BATCH * SEQ)            // 4096
#define SCALING 0.125f                  // HD^-0.5
#define LAMBDA_INIT_F 0.7778700995592559f
#define LN_EPS 1e-5f

typedef __attribute__((ext_vector_type(16))) __bf16 v16bf;
typedef __attribute__((ext_vector_type(8)))  float  v8f;
typedef __attribute__((ext_vector_type(8)))  unsigned short v8us;
typedef __attribute__((ext_vector_type(4)))  unsigned int v4u32;
typedef __attribute__((ext_vector_type(8)))  int    v8i32;
typedef __attribute__((ext_vector_type(4)))  int    v4i32;

// ----------------------------- helpers -------------------------------------

__device__ __forceinline__ __bf16 f2bf(float f) { return (__bf16)f; }

__device__ __forceinline__ __bf16 bits2bf(unsigned short h) {
  union { unsigned short s; __bf16 b; } o; o.s = h; return o.b;
}
__device__ __forceinline__ unsigned short bf2bits(__bf16 b) {
  union { __bf16 b; unsigned short s; } o; o.b = b; return o.s;
}

// DPP-based xor-shuffle within a 16-lane group (enough for the C-layout row
// reductions: masks 1/2/4 are intra-8-lane dpp8 permutes; mask 8 == row_ror:8
// on a 16-lane DPP row).
#define DPP8_SEL(s0,s1,s2,s3,s4,s5,s6,s7) \
  ((s0)|((s1)<<3)|((s2)<<6)|((s3)<<9)|((s4)<<12)|((s5)<<15)|((s6)<<18)|((s7)<<21))

template<int MASK>
__device__ __forceinline__ float xshfl(float x) {
  int i = __builtin_bit_cast(int, x);
  if (MASK == 1)
    i = __builtin_amdgcn_mov_dpp8(i, DPP8_SEL(1,0,3,2,5,4,7,6));
  else if (MASK == 2)
    i = __builtin_amdgcn_mov_dpp8(i, DPP8_SEL(2,3,0,1,6,7,4,5));
  else if (MASK == 4)
    i = __builtin_amdgcn_mov_dpp8(i, DPP8_SEL(4,5,6,7,0,1,2,3));
  else  // MASK == 8: rotate by 8 within the 16-lane row == xor 8
    i = __builtin_amdgcn_mov_dpp(i, 0x128 /*row_ror:8*/, 0xF, 0xF, true);
  return __builtin_bit_cast(float, i);
}

// Build a 16-element bf16 fragment from two contiguous 16B runs (p[0..7],
// q[0..7]); lowers to two b128 loads (LDS or global).
__device__ __forceinline__ v16bf ld_frag2(const unsigned short* p,
                                          const unsigned short* q) {
  union { struct { v8us lo, hi; } s; v16bf v; } u;
  u.s.lo = *(const v8us*)p;
  u.s.hi = *(const v8us*)q;
  return u.v;
}

// A-fragment (16x32 bf16, ISA 7.12.2): lane L holds row M=L&15;
// element i -> K = 8*(L>>4) + (i<8 ? i : 8+i)
__device__ __forceinline__ int a_kidx(int lane, int i) {
  return ((lane >> 4) << 3) + (i < 8 ? i : 8 + i);
}
// B-fragment (32x16 bf16): lane L holds col N=L&15; element i -> K = 8*(L>>4) + (i<8 ? i : 8+i)
__device__ __forceinline__ int b_kidx(int lane, int i) {
  return ((lane >> 4) << 3) + (i < 8 ? i : 8 + i);
}

// A-fragment from a bf16 row in memory: runs at kbase and kbase+16
__device__ __forceinline__ v16bf ld_afrag(const unsigned short* row, int lane) {
  const int kbase = (lane >> 4) << 3;
  return ld_frag2(row + kbase, row + kbase + 16);
}
// B-fragment from a bf16 "column-run" base: runs at kbase and kbase+16
__device__ __forceinline__ v16bf ld_bfrag(const unsigned short* base, int lane) {
  const int kbase = (lane >> 4) << 3;
  return ld_frag2(base + kbase, base + kbase + 16);
}

__device__ __forceinline__ v8f wmma_bf16(v16bf a, v16bf b, v8f c) {
  return __builtin_amdgcn_wmma_f32_16x16x32_bf16(false, a, false, b,
                                                 (short)0, c, false, false);
}

// LDS byte offset of a __shared__ object (flat LDS aperture: addr[31:0])
__device__ __forceinline__ unsigned lds_off(const void* p) {
  return (unsigned)(unsigned long long)(uintptr_t)p;
}

// Tensor Data Mover: DMA a 2D tile (tile1 rows x tile0 elements) from a
// row-major tensor (dim1 x dim0, row stride stride0 elements) into LDS.
// D# layout per CDNA5 ISA ch.8; 6-arg builtin form of this toolchain.
__device__ __forceinline__
void tdm_load_2d(unsigned ldsAddr, const void* gAddr, unsigned dataSizeCode,
                 unsigned tile0, unsigned tile1,
                 unsigned dim0, unsigned dim1, unsigned long long stride0)
{
  const unsigned long long ga = (unsigned long long)(uintptr_t)gAddr;
  v4u32 g0;
  g0.x = 1u;                                               // count=1, user mode
  g0.y = ldsAddr;                                          // lds_addr
  g0.z = (unsigned)ga;                                     // global_addr[31:0]
  g0.w = (unsigned)((ga >> 32) & 0x01FFFFFFu) | (2u << 30);// addr[56:32]|type=2
  v8i32 g1;
  g1[0] = (int)(dataSizeCode << 16);                       // data_size
  g1[1] = (int)((dim0 & 0xFFFFu) << 16);                   // tensor_dim0 lo
  g1[2] = (int)((dim0 >> 16) | ((dim1 & 0xFFFFu) << 16));  // dim0 hi | dim1 lo
  g1[3] = (int)((dim1 >> 16) | (tile0 << 16));             // dim1 hi | tile0
  g1[4] = (int)tile1;                                      // tile1 (tile2=0)
  g1[5] = (int)(stride0 & 0xFFFFFFFFu);                    // dim0_stride lo
  g1[6] = (int)((stride0 >> 32) & 0xFFFFu);                // dim0_stride hi
  g1[7] = 0;
  const v4i32 z4 = {0, 0, 0, 0};
  const v8i32 z8 = {0, 0, 0, 0, 0, 0, 0, 0};
  __builtin_amdgcn_tensor_load_to_lds(g0, g1, z4, z4, z8, 0);
}

// ----------------------------- GEMM ----------------------------------------
// C[M=4096, N] = A[4096,1024] @ W[1024,N]
// MODE 0: A = fp32 (x), store bf16 scattered into [B][H][S][1<<hshift]
// MODE 2: A = bf16 (attn buffer), store fp32 row-major
// Per block: 32x64 output tile, K-slabs of 32 staged into LDS by the TDM
// (double-buffered: slab k+32 DMAs while slab k feeds the WMMAs).
template<int MODE>
__global__ __launch_bounds__(256)
void wmma_gemm(const float* __restrict__ A32,
               const unsigned short* __restrict__ A16,
               const float* __restrict__ W,
               unsigned short* __restrict__ outBf,
               float* __restrict__ outF,
               int N, int hshift)
{
  __shared__ __align__(16) float         shW[2][32 * 64];      // 2 x 8 KB
  __shared__ __align__(16) unsigned char shA[2][32 * 32 * 4];  // 2 x 4 KB

  const int lane  = threadIdx.x & 31;
  const int wave  = threadIdx.x >> 5;
  const int blkN0 = blockIdx.x * 64;
  const int blkM0 = blockIdx.y * 32;
  const int tileN = (wave & 3) * 16;       // local within block tile
  const int tileM = (wave >> 2) * 16;
  const int l15   = lane & 15;

  auto issue = [&](int kk, int buf) {
    // W slab: rows kk..kk+31, cols blkN0..blkN0+63 (f32)
    tdm_load_2d(lds_off(&shW[buf][0]), &W[(size_t)kk * N + blkN0],
                2u, 64u, 32u, (unsigned)N, (unsigned)DIMM,
                (unsigned long long)N);
    // A slab: rows blkM0..blkM0+31, cols kk..kk+31
    if (MODE == 2)
      tdm_load_2d(lds_off(&shA[buf][0]), &A16[(size_t)blkM0 * DIMM + kk],
                  1u, 32u, 32u, (unsigned)DIMM, (unsigned)NROW,
                  (unsigned long long)DIMM);
    else
      tdm_load_2d(lds_off(&shA[buf][0]), &A32[(size_t)blkM0 * DIMM + kk],
                  2u, 32u, 32u, (unsigned)DIMM, (unsigned)NROW,
                  (unsigned long long)DIMM);
  };

  v8f acc = {0.f, 0.f, 0.f, 0.f, 0.f, 0.f, 0.f, 0.f};

  if (wave == 0) issue(0, 0);
  int buf = 0;
  for (int kk = 0; kk < DIMM; kk += 32) {
    if (wave == 0) {
      if (kk + 32 < DIMM) {
        issue(kk + 32, buf ^ 1);                   // prefetch next slab pair
        __builtin_amdgcn_s_wait_tensorcnt(2);      // current pair complete
      } else {
        __builtin_amdgcn_s_wait_tensorcnt(0);
      }
    }
    __syncthreads();

    v16bf a, b;
    const float* wS = &shW[buf][0];
    if (MODE == 2) {
      const unsigned short* aS = (const unsigned short*)&shA[buf][0];
      a = ld_afrag(&aS[(tileM + l15) * 32], lane);      // 2 x ds_load_b128
    } else {
      const float* aS = (const float*)&shA[buf][0];
      const float* row = &aS[(tileM + l15) * 32];
#pragma unroll
      for (int i = 0; i < 16; ++i) a[i] = f2bf(row[a_kidx(lane, i)]);
    }
#pragma unroll
    for (int i = 0; i < 16; ++i) {                       // strided f32 column
      const int kb = b_kidx(lane, i);
      b[i] = f2bf(wS[kb * 64 + tileN + l15]);
    }
    acc = wmma_bf16(a, b, acc);

    __syncthreads();            // all reads done before next DMA overwrites
    buf ^= 1;
  }

  const int colB  = blkN0 + tileN + l15;
  const int mBase = blkM0 + tileM + ((lane >> 4) << 3);
#pragma unroll
  for (int j = 0; j < 8; ++j) {
    const int m = mBase + j;
    if (MODE == 2) {
      outF[(size_t)m * N + colB] = acc[j];
    } else {
      const int bb = m >> 11;                 // batch
      const int s  = m & (SEQ - 1);           // seq pos
      const int h  = colB >> hshift;          // head
      const int dd = colB & ((1 << hshift) - 1);
      const size_t idx =
          ((((size_t)bb * NHEAD + h) * SEQ) + s) * (size_t)(1 << hshift) + dd;
      outBf[idx] = bf2bits(f2bf(acc[j]));
    }
  }
}

// ------------------------ differential flash attention ----------------------

template<int OFF>   // OFF = 0 -> stream 1 (dims 0..63); OFF = 64 -> stream 2
__device__ __forceinline__
void attn_step(const unsigned short* __restrict__ shK,   // [32][128] bf16 slab
               const unsigned short* __restrict__ shV,   // [32][64]  bf16 slab
               int lane, const v16bf (&qa)[2],
               float (&m)[8], float (&l)[8], v8f (&acc)[4],
               unsigned short* __restrict__ pbuf)
{
  const int l15 = lane & 15;
  v8f sLo = {0.f,0.f,0.f,0.f,0.f,0.f,0.f,0.f};
  v8f sHi = sLo;

  // scores: Q (16x64) x K^T (64x16) per 16-key subtile, two k-steps of 32.
  // B-fragment elements are two 8-bf16 runs (k 8h.., 8h+16..) of a shK row -> b128 loads.
#pragma unroll
  for (int f = 0; f < 2; ++f) {
    const v16bf kbLo = ld_bfrag(&shK[(     l15) * 128 + OFF + f * 32], lane);
    const v16bf kbHi = ld_bfrag(&shK[(16 + l15) * 128 + OFF + f * 32], lane);
    sLo = wmma_bf16(qa[f], kbLo, sLo);
    sHi = wmma_bf16(qa[f], kbHi, sHi);
  }

#pragma unroll
  for (int j = 0; j < 8; ++j) { sLo[j] *= SCALING; sHi[j] *= SCALING; }

  // row max over the 32 keys; DPP xor-reduce inside the 16-lane layout group
  float rm[8];
#pragma unroll
  for (int j = 0; j < 8; ++j) {
    float r = fmaxf(sLo[j], sHi[j]);
    r = fmaxf(r, xshfl<1>(r));
    r = fmaxf(r, xshfl<2>(r));
    r = fmaxf(r, xshfl<4>(r));
    r = fmaxf(r, xshfl<8>(r));
    rm[j] = r;
  }

  float alpha[8], rs[8];
#pragma unroll
  for (int j = 0; j < 8; ++j) {
    const float mn = fmaxf(m[j], rm[j]);
    alpha[j] = __expf(m[j] - mn);
    m[j] = mn;
    sLo[j] = __expf(sLo[j] - mn);
    sHi[j] = __expf(sHi[j] - mn);
    rs[j] = sLo[j] + sHi[j];
  }
#pragma unroll
  for (int j = 0; j < 8; ++j) {
    float r = rs[j];
    r += xshfl<1>(r);
    r += xshfl<2>(r);
    r += xshfl<4>(r);
    r += xshfl<8>(r);
    l[j] = l[j] * alpha[j] + r;
  }
#pragma unroll
  for (int t = 0; t < 4; ++t)
#pragma unroll
    for (int j = 0; j < 8; ++j) acc[t][j] *= alpha[j];

  // P (C layout) -> LDS -> A layout for P@V
  const int rbase = (lane >> 4) << 3;
#pragma unroll
  for (int j = 0; j < 8; ++j) {
    pbuf[(rbase + j) * 32 +      l15] = bf2bits(f2bf(sLo[j]));
    pbuf[(rbase + j) * 32 + 16 + l15] = bf2bits(f2bf(sHi[j]));
  }
  asm volatile("s_wait_dscnt 0" ::: "memory");   // wave-local RAW fence on LDS

  const v16bf pa = ld_afrag(&pbuf[l15 * 32], lane);      // 2 x ds_load_b128

#pragma unroll
  for (int t = 0; t < 4; ++t) {
    v16bf vb;                                            // strided across keys
#pragma unroll
    for (int i = 0; i < 16; ++i) {
      const int key = b_kidx(lane, i);
      vb[i] = bits2bf(shV[key * 64 + t * 16 + l15]);
    }
    acc[t] = wmma_bf16(pa, vb, acc[t]);
  }
}

__global__ __launch_bounds__(256)
void diff_attn_kernel(const unsigned short* __restrict__ Q,   // [B][H][S][128]
                      const unsigned short* __restrict__ K,   // [B][H][S][128]
                      const unsigned short* __restrict__ V,   // [B][H][S][64]
                      const float* __restrict__ lq1, const float* __restrict__ lk1,
                      const float* __restrict__ lq2, const float* __restrict__ lk2,
                      unsigned short* __restrict__ Out)       // [B][S][DIM] bf16
{
  __shared__ __align__(16) unsigned short shK[2][32 * 128];  // 2 x 8 KB
  __shared__ __align__(16) unsigned short shV[2][32 * 64];   // 2 x 4 KB
  __shared__ __align__(16) unsigned short Pbuf[8][16 * 32];  // 8 KB

  const int lane = threadIdx.x & 31;
  const int wave = threadIdx.x >> 5;
  const int bh   = blockIdx.x;                  // 0..31 (b*16 + h)
  const int b    = bh >> 4;
  const int h    = bh & 15;
  const int q0   = (blockIdx.y * 8 + wave) * 16;
  const int l15  = lane & 15;

  // lambda scalar (uniform, scalar-cached loads)
  float s1 = 0.f, s2 = 0.f;
  for (int i = 0; i < HD; ++i) { s1 += lq1[i] * lk1[i]; s2 += lq2[i] * lk2[i]; }
  const float lamb = __expf(s1) - __expf(s2) + LAMBDA_INIT_F;

  const size_t qkBase = (size_t)bh * SEQ * 128;
  const size_t vBase  = (size_t)bh * SEQ * 64;

  auto issueKV = [&](int kt, int bf) {
    // K slab: 32 keys x 128 dims (bf16); V slab: 32 keys x 64 dims (bf16)
    tdm_load_2d(lds_off(&shK[bf][0]), &K[qkBase + (size_t)kt * 128],
                1u, 128u, 32u, 128u, (unsigned)SEQ, 128ull);
    tdm_load_2d(lds_off(&shV[bf][0]), &V[vBase + (size_t)kt * 64],
                1u, 64u, 32u, 64u, (unsigned)SEQ, 64ull);
  };

  // load both Q streams as A fragments once (rows M = l15); b128 global loads
  v16bf qa1[2], qa2[2];
  {
    const unsigned short* qrow = &Q[qkBase + (size_t)(q0 + l15) * 128];
#pragma unroll
    for (int f = 0; f < 2; ++f) {
      qa1[f] = ld_afrag(qrow + f * 32, lane);
      qa2[f] = ld_afrag(qrow + 64 + f * 32, lane);
    }
  }

  float m1[8], l1[8], m2[8], l2[8];
  v8f acc1[4], acc2[4];
  const v8f vz = {0.f,0.f,0.f,0.f,0.f,0.f,0.f,0.f};
#pragma unroll
  for (int j = 0; j < 8; ++j) { m1[j] = m2[j] = -1e30f; l1[j] = l2[j] = 0.f; }
#pragma unroll
  for (int t = 0; t < 4; ++t) { acc1[t] = vz; acc2[t] = vz; }

  unsigned short* pbuf = &Pbuf[wave][0];

  if (wave == 0) issueKV(0, 0);
  int buf = 0;
  for (int kt = 0; kt < SEQ; kt += 32) {
    if (wave == 0) {
      if (kt + 32 < SEQ) {
        issueKV(kt + 32, buf ^ 1);                 // DMA next K/V slab pair
        __builtin_amdgcn_s_wait_tensorcnt(2);      // current pair landed
      } else {
        __builtin_amdgcn_s_wait_tensorcnt(0);
      }
    }
    __syncthreads();
    attn_step<0 >(&shK[buf][0], &shV[buf][0], lane, qa1, m1, l1, acc1, pbuf);
    attn_step<64>(&shK[buf][0], &shV[buf][0], lane, qa2, m2, l2, acc2, pbuf);
    __syncthreads();
    buf ^= 1;
  }

  // out = softmax1 - lamb * softmax2, scattered to [B][S][H*64+d] (bf16)
  const int rbase = (lane >> 4) << 3;
#pragma unroll
  for (int t = 0; t < 4; ++t)
#pragma unroll
    for (int j = 0; j < 8; ++j) {
      const float o = acc1[t][j] / l1[j] - lamb * (acc2[t][j] / l2[j]);
      const int srow = q0 + rbase + j;
      const size_t idx = ((size_t)b * SEQ + srow) * DIMM + h * HD + t * 16 + l15;
      Out[idx] = bf2bits(f2bf(o));
    }
}

// ------------------------------- LayerNorm ----------------------------------

__global__ __launch_bounds__(256)
void layernorm_kernel(float* __restrict__ io,
                      const float* __restrict__ gamma,
                      const float* __restrict__ beta)
{
  __shared__ float sh_s[8], sh_ss[8];
  const int row = blockIdx.x;
  float v[4];
  float s = 0.f, ss = 0.f;
#pragma unroll
  for (int i = 0; i < 4; ++i) {
    v[i] = io[(size_t)row * DIMM + i * 256 + threadIdx.x];
    s += v[i]; ss += v[i] * v[i];
  }
#pragma unroll
  for (int mask = 16; mask; mask >>= 1) {
    s  += __shfl_xor(s,  mask, 32);
    ss += __shfl_xor(ss, mask, 32);
  }
  if ((threadIdx.x & 31) == 0) { sh_s[threadIdx.x >> 5] = s; sh_ss[threadIdx.x >> 5] = ss; }
  __syncthreads();
  float ts = 0.f, tss = 0.f;
#pragma unroll
  for (int w = 0; w < 8; ++w) { ts += sh_s[w]; tss += sh_ss[w]; }
  const float mu  = ts * (1.0f / DIMM);
  const float var = tss * (1.0f / DIMM) - mu * mu;
  const float inv = rsqrtf(var + LN_EPS);
#pragma unroll
  for (int i = 0; i < 4; ++i) {
    const int c = i * 256 + threadIdx.x;
    io[(size_t)row * DIMM + c] = (v[i] - mu) * inv * gamma[c] + beta[c];
  }
}

// ------------------------------- launcher -----------------------------------

extern "C" void kernel_launch(void* const* d_in, const int* in_sizes, int n_in,
                              void* d_out, int out_size, void* d_ws, size_t ws_size,
                              hipStream_t stream) {
  const float* x   = (const float*)d_in[0];
  const float* Wq  = (const float*)d_in[1];
  const float* Wk  = (const float*)d_in[2];
  const float* Wv  = (const float*)d_in[3];
  const float* Wo  = (const float*)d_in[4];
  const float* lq1 = (const float*)d_in[5];
  const float* lk1 = (const float*)d_in[6];
  const float* lq2 = (const float*)d_in[7];
  const float* lk2 = (const float*)d_in[8];
  const float* gamma = (const float*)d_in[9];
  const float* beta  = (const float*)d_in[10];
  float* out = (float*)d_out;

  // workspace: bf16 staging buffers (48 MB total)
  unsigned short* q_ws = (unsigned short*)d_ws;                       // [B][H][S][128]
  unsigned short* k_ws = q_ws + (size_t)NROW * 2048;                  // [B][H][S][128]
  unsigned short* v_ws = k_ws + (size_t)NROW * 2048;                  // [B][H][S][64]
  unsigned short* a_ws = v_ws + (size_t)NROW * 1024;                  // [B][S][1024]

  const dim3 blk(256);
  // projections
  wmma_gemm<0><<<dim3(32, 128), blk, 0, stream>>>(x, nullptr, Wq, q_ws, nullptr, 2048, 7);
  wmma_gemm<0><<<dim3(32, 128), blk, 0, stream>>>(x, nullptr, Wk, k_ws, nullptr, 2048, 7);
  wmma_gemm<0><<<dim3(16, 128), blk, 0, stream>>>(x, nullptr, Wv, v_ws, nullptr, 1024, 6);
  // differential attention
  diff_attn_kernel<<<dim3(32, 16), blk, 0, stream>>>(q_ws, k_ws, v_ws,
                                                     lq1, lk1, lq2, lk2, a_ws);
  // output projection (fp32 result straight into d_out)
  wmma_gemm<2><<<dim3(16, 128), blk, 0, stream>>>(nullptr, a_ws, Wo, nullptr, out, 1024, 0);
  // in-place LayerNorm
  layernorm_kernel<<<dim3(NROW), blk, 0, stream>>>(out, gamma, beta);
}
